// Attention_65420941852829
// MI455X (gfx1250) — matrix operand order repaired and checked by reading, not compiled, not run on hardware
//
#include <hip/hip_runtime.h>
#include <hip/hip_fp16.h>

// ---------------------------------------------------------------------------
// MHA forward for MI455X (gfx1250), wave32 + WMMA f16 (16x16x32, f32 accum).
//   qp = (q*SCALE) @ Wq^T + bq ; kp = k @ Wk^T + bk ; vp = v @ Wv^T + bv
//   per head: softmax(qh kh^T) vh ; out = attn @ Wp^T + bp
// All GEMMs run on v_wmma_f32_16x16x32_f16. K/V tiles staged to LDS with
// global_load_async_to_lds_b128 + s_wait_asynccnt when the toolchain has the
// builtins (gated by __has_builtin, manual copy otherwise).
// ---------------------------------------------------------------------------

typedef _Float16 half_t;
typedef __attribute__((ext_vector_type(16))) _Float16 v16h;
typedef __attribute__((ext_vector_type(8)))  float    v8f;

#define T_LEN  2048
#define BSZ    4
#define DIM    256
#define NH     8
#define HD     32
#define ROWS   (T_LEN * BSZ)          // 8192
#define QK_SCALE 0.17677669529663687f // 32^-0.5
#define SCHUNK 64

#if defined(__gfx1250__) &&                                              \
    __has_builtin(__builtin_amdgcn_global_load_async_to_lds_b128) &&     \
    __has_builtin(__builtin_amdgcn_s_wait_asynccnt)
#define USE_ASYNC_LDS 1
// Signature (from hipcc diagnostic): arg0 is `int __vector(4)*` (global side).
typedef int b128_t __attribute__((vector_size(16)));
typedef __attribute__((address_space(1))) b128_t* gptr128_t;
typedef __attribute__((address_space(3))) b128_t* lptr128_t;
#else
#define USE_ASYNC_LDS 0
#endif

union F16x16 {
    v16h         v;
    _Float16     h[16];
    unsigned int u[8];
};

__device__ __forceinline__ v8f wmma_f16(const F16x16& a, const F16x16& b, v8f c) {
    // (neg_a, A, neg_b, B, c_mod, C, reuse_a, reuse_b)
    return __builtin_amdgcn_wmma_f32_16x16x32_f16(false, a.v, false, b.v,
                                                  (short)0, c, false, false);
}

// ---------------------------------------------------------------------------
// One-shot fp32 -> f16 conversion of the four 256x256 weight matrices.
// Wh layout: [Wq | Wk | Wv | Wp], each row-major 256x256.
// ---------------------------------------------------------------------------
__global__ __launch_bounds__(256)
void convert_weights(const float* __restrict__ W0, const float* __restrict__ W1,
                     const float* __restrict__ W2, const float* __restrict__ W3,
                     half_t* __restrict__ Wh)
{
    const int i = blockIdx.x * 256 + threadIdx.x;  // float2 index, 0..32767
    const float* src[4] = {W0, W1, W2, W3};
    #pragma unroll
    for (int m = 0; m < 4; ++m) {
        const float2 x = ((const float2*)src[m])[i];
        half_t* dst = Wh + (size_t)m * (DIM * DIM) + 2 * i;
        dst[0] = (half_t)x.x;
        dst[1] = (half_t)x.y;
    }
}

// ---------------------------------------------------------------------------
// Projection: Y_heads[(b*8+h)][t][d] = f16( X[r=t*4+b, :] (*alpha) @ W^T + b )
// Block = 256 threads = 8 waves; each wave owns 16 rows; block tile = 128 rows.
// A fragment (16x32 f16): lane m = L&15, k-half = (L>>4)*8, 2 K packed / VGPR.
// B fragment (32x16 f16): lane n = L&15, k-base = (L>>4)*16, K = kb+2v, kb+2v+1
//                         => B[k][n] = W[n][k] is contiguous in W rows.
// C/D (16x16 f32): row = r + 8*(L>>4), col = L&15.
// ---------------------------------------------------------------------------
__global__ __launch_bounds__(256)
void proj_to_heads(const float* __restrict__ X, const half_t* __restrict__ Wh,
                   const float* __restrict__ bias, half_t* __restrict__ Y,
                   float alpha)
{
    const int lane  = threadIdx.x & 31;
    const int wave  = threadIdx.x >> 5;
    const int rbase = blockIdx.x * 128 + wave * 16;
    const int m     = lane & 15;
    const int kh    = (lane >> 4) * 8;
    const int r     = rbase + m;

    // Preload all 8 K-step A-fragments for this wave's 16 rows.
    F16x16 a[8];
    const float2* Xrow = (const float2*)(X + (size_t)r * DIM);
    #pragma unroll
    for (int ks = 0; ks < 8; ++ks) {
        #pragma unroll
        for (int v = 0; v < 8; ++v) {
            const int k0 = ks * 32 + 2 * kh + 2 * v;
            const float2 x2 = Xrow[k0 >> 1];
            a[ks].h[2 * v]     = (_Float16)(alpha * x2.x);
            a[ks].h[2 * v + 1] = (_Float16)(alpha * x2.y);
        }
    }

    const int nb = lane & 15;
    const int kb = (lane >> 4) * 16;

    for (int nt = 0; nt < 16; ++nt) {
        const int n = nt * 16 + nb;
        const unsigned int* Wrow = (const unsigned int*)(Wh + (size_t)n * DIM);
        v8f c = {};
        #pragma unroll
        for (int ks = 0; ks < 8; ++ks) {
            F16x16 b;
            const int base = (ks * 32 + kb) >> 1;  // dword index, K contiguous
            #pragma unroll
            for (int v = 0; v < 8; ++v) b.u[v] = Wrow[base + v];
            c = wmma_f16(a[ks], b, c);
        }
        const float bn = bias[n];
        const int h = n >> 5, d = n & 31;
        #pragma unroll
        for (int rr = 0; rr < 8; ++rr) {
            const int row = rbase + rr + 8 * (lane >> 4);
            const int t = row >> 2, bb = row & 3;
            Y[((size_t)(bb * NH + h) * T_LEN + t) * HD + d] =
                (half_t)(c[rr] + bn);
        }
    }
}

// ---------------------------------------------------------------------------
// Flash attention. Grid: (T/128, B*H). Block = 256 thr = 8 waves, wave = 16 q.
// ---------------------------------------------------------------------------
__global__ __launch_bounds__(256)
void attention(const half_t* __restrict__ Qh, const half_t* __restrict__ Kh,
               const half_t* __restrict__ Vh, half_t* __restrict__ Ah)
{
    __shared__ half_t Kc[SCHUNK * HD];       // 4 KB
    __shared__ half_t Vc[SCHUNK * HD];       // 4 KB
    __shared__ half_t Pst[8][16 * SCHUNK];   // 16 KB: per-wave P staging

    const int lane  = threadIdx.x & 31;
    const int wave  = threadIdx.x >> 5;
    const int bh    = blockIdx.y;
    const int tbase = blockIdx.x * 128 + wave * 16;

    const size_t headoff = (size_t)bh * T_LEN * HD;

    // Q A-fragment (16 rows x 32 head-dim) held in registers for whole kernel.
    F16x16 qa;
    {
        const int m  = lane & 15;
        const int kh = (lane >> 4) * 8;
        const unsigned int* Qrow =
            (const unsigned int*)(Qh + headoff + (size_t)(tbase + m) * HD);
        #pragma unroll
        for (int v = 0; v < 8; ++v) {
            const int k0 = 2 * kh + 2 * v;
            qa.u[v] = Qrow[k0 >> 1];
        }
    }

    v8f o0 = {}, o1 = {};
    float mrun[8], lrun[8];
    #pragma unroll
    for (int i = 0; i < 8; ++i) { mrun[i] = -1e30f; lrun[i] = 0.0f; }

    const char* Ksrc = (const char*)(Kh + headoff);
    const char* Vsrc = (const char*)(Vh + headoff);

    const int nb = lane & 15;
    const int kb = (lane >> 4) * 16;

    for (int s0 = 0; s0 < T_LEN; s0 += SCHUNK) {
        __syncthreads();  // previous iteration done with Kc/Vc
        {
            const size_t cb = (size_t)s0 * HD * sizeof(half_t);  // chunk byte base
#if USE_ASYNC_LDS
            // 256 threads x 16B = 4KB = one SCHUNK x HD f16 tile per buffer.
            const size_t off = cb + (size_t)threadIdx.x * 16;
            __builtin_amdgcn_global_load_async_to_lds_b128(
                (gptr128_t)(Ksrc + off),
                (lptr128_t)((char*)Kc + threadIdx.x * 16),
                0, 0);
            __builtin_amdgcn_global_load_async_to_lds_b128(
                (gptr128_t)(Vsrc + off),
                (lptr128_t)((char*)Vc + threadIdx.x * 16),
                0, 0);
            __builtin_amdgcn_s_wait_asynccnt(0);
#else
            const unsigned int* Kw = (const unsigned int*)(Ksrc + cb);
            const unsigned int* Vw = (const unsigned int*)(Vsrc + cb);
            unsigned int* KcW = (unsigned int*)Kc;
            unsigned int* VcW = (unsigned int*)Vc;
            #pragma unroll
            for (int i = 0; i < (SCHUNK * HD / 2) / 256; ++i) {
                const int w = threadIdx.x + i * 256;
                KcW[w] = Kw[w];
                VcW[w] = Vw[w];
            }
#endif
            if (s0 + SCHUNK < T_LEN) {  // prefetch next chunk -> global_prefetch_b8
                __builtin_prefetch(Ksrc + cb + SCHUNK * HD * 2 + threadIdx.x * 16, 0, 1);
                __builtin_prefetch(Vsrc + cb + SCHUNK * HD * 2 + threadIdx.x * 16, 0, 1);
            }
        }
        __syncthreads();

        // ---- S = Q @ Kc^T : 4 tiles of 16 keys (single K-step, head_dim=32)
        v8f s[4];
        #pragma unroll
        for (int st = 0; st < 4; ++st) {
            F16x16 b;
            const unsigned int* Krow =
                (const unsigned int*)(Kc + (st * 16 + nb) * HD);
            #pragma unroll
            for (int v = 0; v < 8; ++v) b.u[v] = Krow[(kb + 2 * v) >> 1];
            v8f z = {};
            s[st] = wmma_f16(qa, b, z);
        }

        // ---- online softmax (rows live in 16-lane halves of the wave)
        float scl[8];
        #pragma unroll
        for (int rr = 0; rr < 8; ++rr) {
            float mx = fmaxf(fmaxf(s[0][rr], s[1][rr]), fmaxf(s[2][rr], s[3][rr]));
            #pragma unroll
            for (int off = 8; off >= 1; off >>= 1)
                mx = fmaxf(mx, __shfl_xor(mx, off, 32));
            const float mnew = fmaxf(mrun[rr], mx);
            scl[rr] = __expf(mrun[rr] - mnew);
            mrun[rr] = mnew;
            float rs = 0.0f;
            #pragma unroll
            for (int st = 0; st < 4; ++st) {
                const float p = __expf(s[st][rr] - mnew);
                s[st][rr] = p;
                rs += p;
            }
            #pragma unroll
            for (int off = 8; off >= 1; off >>= 1)
                rs += __shfl_xor(rs, off, 32);
            lrun[rr] = lrun[rr] * scl[rr] + rs;
            o0[rr] *= scl[rr];
            o1[rr] *= scl[rr];
        }

        // ---- stage P (f16) to LDS, C-layout -> row-major 16 x SCHUNK
        half_t* pw = Pst[wave];
        {
            const int colb   = lane & 15;
            const int rbase8 = 8 * (lane >> 4);
            #pragma unroll
            for (int st = 0; st < 4; ++st)
                #pragma unroll
                for (int rr = 0; rr < 8; ++rr)
                    pw[(rbase8 + rr) * SCHUNK + st * 16 + colb] =
                        (half_t)s[st][rr];
        }
        __syncthreads();  // order P stores before P loads (uniform barrier)

        // ---- O += P @ Vc  (K = 64 -> 2 K-steps; N = 32 -> 2 tiles)
        {
            const int m   = lane & 15;
            const int khh = (lane >> 4) * 8;
            #pragma unroll
            for (int kstep = 0; kstep < 2; ++kstep) {
                F16x16 pa;
                const unsigned int* Prow =
                    (const unsigned int*)(pw + m * SCHUNK + kstep * 32);
                #pragma unroll
                for (int v = 0; v < 8; ++v) {
                    const int k0 = 2 * khh + 2 * v;
                    pa.u[v] = Prow[k0 >> 1];
                }
                #pragma unroll
                for (int j = 0; j < 2; ++j) {
                    F16x16 bv;
                    #pragma unroll
                    for (int v = 0; v < 8; ++v) {
                        const int kk = kstep * 32 + kb + 2 * v;
                        bv.h[2 * v]     = Vc[kk * HD + j * 16 + nb];
                        bv.h[2 * v + 1] = Vc[(kk + 1) * HD + j * 16 + nb];
                    }
                    if (j == 0) o0 = wmma_f16(pa, bv, o0);
                    else        o1 = wmma_f16(pa, bv, o1);
                }
            }
        }
    }

    // ---- normalize and store to [t, b, h*32 + d] (f16) for final projection
    const int h = bh & 7, bb = bh >> 3;
    #pragma unroll
    for (int rr = 0; rr < 8; ++rr) {
        const float inv = 1.0f / lrun[rr];
        const int t = tbase + rr + 8 * (lane >> 4);
        const size_t base = ((size_t)(t * BSZ + bb)) * DIM + h * HD;
        Ah[base + (lane & 15)]      = (half_t)(o0[rr] * inv);
        Ah[base + 16 + (lane & 15)] = (half_t)(o1[rr] * inv);
    }
}

// ---------------------------------------------------------------------------
// Output projection: out[r, :] = Ah[r, :] @ Wp^T + bp   (f16 in, f32 out)
// ---------------------------------------------------------------------------
__global__ __launch_bounds__(256)
void proj_out(const half_t* __restrict__ X, const half_t* __restrict__ Wh,
              const float* __restrict__ bias, float* __restrict__ Y)
{
    const int lane  = threadIdx.x & 31;
    const int wave  = threadIdx.x >> 5;
    const int rbase = blockIdx.x * 128 + wave * 16;
    const int m     = lane & 15;
    const int kh    = (lane >> 4) * 8;
    const int r     = rbase + m;

    F16x16 a[8];
    const unsigned int* Xrow = (const unsigned int*)(X + (size_t)r * DIM);
    #pragma unroll
    for (int ks = 0; ks < 8; ++ks) {
        #pragma unroll
        for (int v = 0; v < 8; ++v) {
            const int k0 = ks * 32 + 2 * kh + 2 * v;
            a[ks].u[v] = Xrow[k0 >> 1];
        }
    }

    const int nb = lane & 15;
    const int kb = (lane >> 4) * 16;

    for (int nt = 0; nt < 16; ++nt) {
        const int n = nt * 16 + nb;
        const unsigned int* Wrow = (const unsigned int*)(Wh + (size_t)n * DIM);
        v8f c = {};
        #pragma unroll
        for (int ks = 0; ks < 8; ++ks) {
            F16x16 b;
            const int base = (ks * 32 + kb) >> 1;
            #pragma unroll
            for (int v = 0; v < 8; ++v) b.u[v] = Wrow[base + v];
            c = wmma_f16(a[ks], b, c);
        }
        const float bn = bias[n];
        #pragma unroll
        for (int rr = 0; rr < 8; ++rr) {
            const int row = rbase + rr + 8 * (lane >> 4);
            Y[(size_t)row * DIM + n] = c[rr] + bn;
        }
    }
}

// ---------------------------------------------------------------------------
extern "C" void kernel_launch(void* const* d_in, const int* in_sizes, int n_in,
                              void* d_out, int out_size, void* d_ws, size_t ws_size,
                              hipStream_t stream)
{
    const float* q  = (const float*)d_in[0];
    const float* k  = (const float*)d_in[1];
    const float* v  = (const float*)d_in[2];
    const float* Wq = (const float*)d_in[3];
    const float* bq = (const float*)d_in[4];
    const float* Wk = (const float*)d_in[5];
    const float* bk = (const float*)d_in[6];
    const float* Wv = (const float*)d_in[7];
    const float* bv = (const float*)d_in[8];
    const float* Wp = (const float*)d_in[9];
    const float* bp = (const float*)d_in[10];
    float* out = (float*)d_out;

    // Workspace (f16): Qh/Kh/Vh per-head [32][2048][32], Ah [8192][256],
    // Wh = 4 x 256x256 converted weights.  Total ~16.5 MB.
    const size_t headElems = (size_t)BSZ * NH * T_LEN * HD;  // 2,097,152
    half_t* Qh = (half_t*)d_ws;
    half_t* Kh = Qh + headElems;
    half_t* Vh = Kh + headElems;
    half_t* Ah = Vh + headElems;
    half_t* Wh = Ah + (size_t)ROWS * DIM;
    half_t* Wqh = Wh + 0 * (size_t)DIM * DIM;
    half_t* Wkh = Wh + 1 * (size_t)DIM * DIM;
    half_t* Wvh = Wh + 2 * (size_t)DIM * DIM;
    half_t* Wph = Wh + 3 * (size_t)DIM * DIM;

    convert_weights<<<(DIM * DIM / 2) / 256, 256, 0, stream>>>(Wq, Wk, Wv, Wp, Wh);
    proj_to_heads<<<ROWS / 128, 256, 0, stream>>>(q, Wqh, bq, Qh, QK_SCALE);
    proj_to_heads<<<ROWS / 128, 256, 0, stream>>>(k, Wkh, bk, Kh, 1.0f);
    proj_to_heads<<<ROWS / 128, 256, 0, stream>>>(v, Wvh, bv, Vh, 1.0f);
    attention<<<dim3(T_LEN / 128, BSZ * NH), 256, 0, stream>>>(Qh, Kh, Vh, Ah);
    proj_out<<<ROWS / 128, 256, 0, stream>>>(Ah, Wph, bp, out);
}
